// SelfAttn_77747497992398
// MI455X (gfx1250) — matrix operand order repaired and checked by reading, not compiled, not run on hardware
//
#include <hip/hip_runtime.h>

// ---------------------------------------------------------------------------
// SAGAN self-attention, B=4, W=H=64, C=256  —  MI455X (gfx1250), wave32.
// f16 WMMA flash attention + TDM (tensor_load_to_lds) K/V staging in LDS.
// ---------------------------------------------------------------------------

typedef _Float16 h16 __attribute__((ext_vector_type(16)));
typedef _Float16 h8  __attribute__((ext_vector_type(8)));
typedef float    f8  __attribute__((ext_vector_type(8)));
typedef float    f4  __attribute__((ext_vector_type(4)));
typedef unsigned int u32x4 __attribute__((ext_vector_type(4)));
typedef int          i32x4 __attribute__((ext_vector_type(4)));
typedef int          i32x8 __attribute__((ext_vector_type(8)));

#define NTOK 4096   // W*H
#define CIN  256    // C
#define CH   128    // C/2
#define NB   4      // batch

#define HAVE_TDM __has_builtin(__builtin_amdgcn_tensor_load_to_lds)
#define HAVE_TCNT __has_builtin(__builtin_amdgcn_s_wait_tensorcnt)

__device__ __forceinline__ f8 wmma16(h16 a, h16 b, f8 c) {
  return __builtin_amdgcn_wmma_f32_16x16x32_f16(false, a, false, b, (short)0, c,
                                                false, false);
}

// ---------------------------------------------------------------------------
// Prep: f32 -> f16 bulk convert (8 elements / thread)
// ---------------------------------------------------------------------------
__global__ void __launch_bounds__(256)
cvt_kernel(const float* __restrict__ src, _Float16* __restrict__ dst, int n8) {
  const int i = blockIdx.x * blockDim.x + threadIdx.x;
  if (i >= n8) return;
  const f4* s = (const f4*)src + (size_t)i * 2;
  const f4 a = s[0], b = s[1];
  h8 o;
#pragma unroll
  for (int e = 0; e < 4; ++e) { o[e] = (_Float16)a[e]; o[e + 4] = (_Float16)b[e]; }
  *((h8*)dst + i) = o;
}

// Prep: WT[j][k] = (f16) W[k][j]   (k < CIN=256, j < Cout)
__global__ void __launch_bounds__(256)
wtr_kernel(const float* __restrict__ W, _Float16* __restrict__ WT, int Cout) {
  const int idx = blockIdx.x * blockDim.x + threadIdx.x;
  if (idx >= Cout * CIN) return;
  const int j = idx >> 8;    // / CIN
  const int k = idx & 255;   // % CIN
  WT[idx] = (_Float16)W[k * Cout + j];
}

// ---------------------------------------------------------------------------
// Projection: Y[n][j] = sum_c Xh[n][c]*WT[j][c] + bias[j]   (f16 out)
// layout 0: raw [N][Cout] (v; raw buffer == V2[C][N] view)
// layout 1: scramble-transposed: dst = ((n&31)*128 + j)*128 + (n>>5)
// One wave = one 16x16 tile; all fragment loads contiguous f16.
// ---------------------------------------------------------------------------
__global__ void __launch_bounds__(256)
proj_kernel(const _Float16* __restrict__ Xh, const _Float16* __restrict__ WT,
            const float* __restrict__ bias, _Float16* __restrict__ Y,
            int Cout, int layout) {
  const int lane = threadIdx.x & 31;
  const int gw   = blockIdx.x * (blockDim.x >> 5) + (threadIdx.x >> 5);

  const int tilesJ = Cout >> 4;
  const int wavesPerBatch = (NTOK >> 4) * tilesJ;
  const int b  = gw / wavesPerBatch;
  const int r  = gw % wavesPerBatch;
  const int n0 = (r / tilesJ) << 4;
  const int j0 = (r % tilesJ) << 4;

  const _Float16* Xb = Xh + (size_t)b * NTOK * CIN;
  _Float16*       Yb = Y  + (size_t)b * NTOK * Cout;

  const int lm = lane & 15;
  const int hh = lane >> 4;
  const _Float16* arow = Xb + (size_t)(n0 + lm) * CIN;
  const _Float16* brow = WT + (size_t)(j0 + lm) * CIN;

  f8 acc = {};
#pragma unroll
  for (int k0 = 0; k0 < CIN; k0 += 32) {
    h16 a;
    const h8 lo = *(const h8*)(arow + k0 + 16 * hh);
    const h8 hi = *(const h8*)(arow + k0 + 16 * hh + 8);
#pragma unroll
    for (int e = 0; e < 8; ++e) { a[e] = lo[e]; a[e + 8] = hi[e]; }
    const h16 bf = *(const h16*)(brow + k0 + 16 * hh);
    acc = wmma16(a, bf, acc);
  }

  const int j = j0 + lm;
  const float bj = bias[j];
#pragma unroll
  for (int rr = 0; rr < 8; ++rr) {
    const int n = n0 + rr + 8 * hh;
    const _Float16 val = (_Float16)(acc[rr] + bj);
    size_t dst;
    if (layout == 0) dst = (size_t)n * Cout + j;
    else             dst = ((size_t)(n & 31) * 128 + j) * 128 + (size_t)(n >> 5);
    Yb[dst] = val;
  }
}

// ---------------------------------------------------------------------------
// LDS carve (single dynamic region; offsets double as TDM lds_addr values)
// ---------------------------------------------------------------------------
#define K_OFF   0        // Kl[2][32][128] f16  : 2 * 8192 B
#define V_OFF   16384    // Vl[2][256][32] f16  : 2 * 16384 B
#define S_OFF   49152    // Sld[8][16*32] f32   : 16384 B
#define P_OFF   65536    // Pld[8][16*32] f16   : 8192 B
#define SC_OFF  73728    // scale[8][16] f32    : 512 B
#define L_OFF   74240    // l[8][16] f32        : 512 B
#define SMEM_BYTES 74752

#if HAVE_TDM
// 2-D TDM load: tile (tileW x tileH) of 2-byte elements, global row stride
// strideElems, into contiguous LDS at byte offset ldsOff.  D# per ISA 8.3-8.6.
__device__ __forceinline__ void tdm_load_2d(unsigned ldsOff, const _Float16* gptr,
                                            unsigned tileW, unsigned tileH,
                                            unsigned long long strideElems) {
  const unsigned long long ga = (unsigned long long)gptr;
  u32x4 g0;
  g0[0] = 1u;                                   // count=1 (valid), user mode
  g0[1] = ldsOff;                               // lds_addr (bytes)
  g0[2] = (unsigned)ga;                         // global_addr[31:0]
  g0[3] = (unsigned)((ga >> 32) & 0x1FFFFFFull) | (2u << 30); // addr[56:32]|type=2
  i32x8 g1;
  g1[0] = (int)(1u << 16);                      // data_size=1 -> 2 bytes
  g1[1] = (int)((tileW & 0xFFFFu) << 16);       // tensor_dim0[15:0]
  g1[2] = (int)(((tileW >> 16) & 0xFFFFu) | ((tileH & 0xFFFFu) << 16));
  g1[3] = (int)(((tileH >> 16) & 0xFFFFu) | ((tileW & 0xFFFFu) << 16)); // tile_dim0
  g1[4] = (int)(tileH & 0xFFFFu);               // tile_dim1 (tile_dim2=0)
  g1[5] = (int)(strideElems & 0xFFFFFFFFull);   // tensor_dim0_stride[31:0]
  g1[6] = (int)((strideElems >> 32) & 0xFFFFull);
  g1[7] = 0;
  const i32x4 z4 = {0, 0, 0, 0};
#if defined(__clang_major__) && (__clang_major__ >= 23)
  const i32x8 z8 = {0, 0, 0, 0, 0, 0, 0, 0};
  __builtin_amdgcn_tensor_load_to_lds(g0, g1, z4, z4, z8, 0);
#else
  __builtin_amdgcn_tensor_load_to_lds(g0, g1, z4, z4, 0);
#endif
}

template <int N>
__device__ __forceinline__ void wait_tensorcnt() {
#if HAVE_TCNT
  __builtin_amdgcn_s_wait_tensorcnt((short)N);  // N is an ICE (template param)
#else
  asm volatile("s_wait_tensorcnt %0" :: "i"(N) : "memory");
#endif
}
#endif

// ---------------------------------------------------------------------------
// Flash attention.  One wave = 16 query rows x all 256 channels.
// 8 waves/block share TDM-staged, double-buffered K/V chunks (32 keys/iter).
// ---------------------------------------------------------------------------
#define WPB 8
__global__ void __launch_bounds__(256)
attn_kernel(const _Float16* __restrict__ QT, const _Float16* __restrict__ KT,
            const _Float16* __restrict__ V2, const float* __restrict__ x,
            const float* __restrict__ gamma, float* __restrict__ out) {
  extern __shared__ char smem[];
  _Float16* Kl = (_Float16*)(smem + K_OFF);
  _Float16* Vl = (_Float16*)(smem + V_OFF);
  const int wv   = threadIdx.x >> 5;
  const int lane = threadIdx.x & 31;
  float*    Sw  = (float*)(smem + S_OFF) + wv * 512;
  _Float16* Pw  = (_Float16*)(smem + P_OFF) + wv * 512;
  float*    scW = (float*)(smem + SC_OFF) + wv * 16;
  float*    lW  = (float*)(smem + L_OFF) + wv * 16;

  const int gw = blockIdx.x * WPB + wv;
  const int b  = gw >> 8;
  const int n0 = (gw & 255) << 4;

  const _Float16* QTb = QT + (size_t)b * NTOK * CH;
  const _Float16* KTb = KT + (size_t)b * NTOK * CH;
  const _Float16* Vb  = V2 + (size_t)b * NTOK * CIN;

  const int lm = lane & 15;
  const int hh = lane >> 4;

  // Q A-fragments for this wave's 16 rows, kept in registers (K = 128).
  h16 qa[4];
  {
    const _Float16* qrow = QTb + (size_t)(n0 + lm) * CH;
#pragma unroll
    for (int kk = 0; kk < 4; ++kk) {
      const h8 lo = *(const h8*)(qrow + kk * 32 + 16 * hh);
      const h8 hi = *(const h8*)(qrow + kk * 32 + 16 * hh + 8);
#pragma unroll
      for (int e = 0; e < 8; ++e) { qa[kk][e] = lo[e]; qa[kk][e + 8] = hi[e]; }
    }
  }

  f8 o[16];
#pragma unroll
  for (int t = 0; t < 16; ++t) o[t] = (f8){};

  float mReg = -3.0e38f, lReg = 0.0f;  // valid in lanes 0-15

#if HAVE_TDM
  if (wv == 0) {   // prologue: chunk 0 (EXEC ignored by TDM; wave-uniform args)
    tdm_load_2d(K_OFF, KTb, CH, 32, CH);
    tdm_load_2d(V_OFF, Vb, 32, CIN, NTOK);
  }
#endif

  const int NCHUNK = NTOK / 32;  // 128
  for (int ch = 0; ch < NCHUNK; ++ch) {
    const int m0  = ch * 32;
    const int cur = ch & 1;
    __syncthreads();  // close previous iteration's K/V/P reads

#if HAVE_TDM
    if (wv == 0) {
      if (ch + 1 < NCHUNK) {
        const unsigned nb = (unsigned)((ch + 1) & 1);
        tdm_load_2d(K_OFF + nb * 8192u,  KTb + (size_t)(m0 + 32) * CH, CH, 32, CH);
        tdm_load_2d(V_OFF + nb * 16384u, Vb + (m0 + 32), 32, CIN, NTOK);
        wait_tensorcnt<2>();  // two oldest (chunk ch) complete; ch+1 in flight
      } else {
        wait_tensorcnt<0>();
      }
    }
#else
    {  // cooperative fallback staging (vector loads, 256 threads)
      const int tid = threadIdx.x;
      *((h16*)(Kl + cur * 4096) + tid) = *((const h16*)(KTb + (size_t)m0 * CH) + tid);
      const _Float16* vsrc = Vb + (size_t)tid * NTOK + m0;
      _Float16*       vdst = Vl + cur * 8192 + tid * 32;
      *(h16*)vdst        = *(const h16*)vsrc;
      *(h16*)(vdst + 16) = *(const h16*)(vsrc + 16);
    }
#endif
    __syncthreads();  // chunk `ch` visible to all waves

    // ---- S = Q K^T for 32 keys (two 16x16 tiles, K-dim 128) ----
    f8 s0 = {}, s1 = {};
    const _Float16* kcur = Kl + (size_t)cur * 4096;
    const _Float16* k0p  = kcur + lm * CH + 16 * hh;
    const _Float16* k1p  = k0p + 16 * CH;
#pragma unroll
    for (int kk = 0; kk < 4; ++kk) {
      const h16 kb0 = *(const h16*)(k0p + kk * 32);
      const h16 kb1 = *(const h16*)(k1p + kk * 32);
      s0 = wmma16(qa[kk], kb0, s0);
      s1 = wmma16(qa[kk], kb1, s1);
    }

    // ---- D-frag -> LDS ----
#pragma unroll
    for (int rr = 0; rr < 8; ++rr) {
      Sw[(rr + 8 * hh) * 32 + lm]      = s0[rr];
      Sw[(rr + 8 * hh) * 32 + 16 + lm] = s1[rr];
    }
    __syncthreads();

    // ---- online softmax: lane L<16 owns query row L ----
    if (lane < 16) {
      const float* srow = Sw + lane * 32;
      float mx = mReg;
#pragma unroll
      for (int jj = 0; jj < 32; ++jj) mx = fmaxf(mx, srow[jj]);
      const float sc = __expf(mReg - mx);
      float sum = 0.0f;
      _Float16* prow = Pw + lane * 32;
#pragma unroll
      for (int jj = 0; jj < 32; ++jj) {
        const float p = __expf(srow[jj] - mx);
        prow[jj] = (_Float16)p;
        sum += p;
      }
      lReg = lReg * sc + sum;
      mReg = mx;
      scW[lane] = sc;
    }
    __syncthreads();

    // ---- rescale O, then O += P @ V^T over 16 channel tiles ----
    float scv[8];
#pragma unroll
    for (int rr = 0; rr < 8; ++rr) scv[rr] = scW[rr + 8 * hh];
#pragma unroll
    for (int t = 0; t < 16; ++t)
#pragma unroll
      for (int rr = 0; rr < 8; ++rr) o[t][rr] *= scv[rr];

    h16 pf;
    {
      const _Float16* prow = Pw + lm * 32;
      const h8 lo = *(const h8*)(prow + 16 * hh);
      const h8 hi = *(const h8*)(prow + 16 * hh + 8);
#pragma unroll
      for (int e = 0; e < 8; ++e) { pf[e] = lo[e]; pf[e + 8] = hi[e]; }
    }
    const _Float16* vcur = Vl + (size_t)cur * 8192;
#pragma unroll
    for (int t = 0; t < 16; ++t) {
      const int c = t * 16 + lm;
      const h16 vf = *(const h16*)(vcur + c * 32 + 16 * hh);
      o[t] = wmma16(pf, vf, o[t]);
    }
  }

  if (lane < 16) lW[lane] = lReg;
  __syncthreads();

  float linv[8];
#pragma unroll
  for (int rr = 0; rr < 8; ++rr) linv[rr] = 1.0f / lW[rr + 8 * hh];

  const float g = gamma[0];
  const float* xb = x   + (size_t)b * NTOK * CIN;
  float*       ob = out + (size_t)b * NTOK * CIN;
#pragma unroll
  for (int t = 0; t < 16; ++t) {
    const int c = t * 16 + lm;
#pragma unroll
    for (int rr = 0; rr < 8; ++rr) {
      const int n = n0 + rr + 8 * hh;
      const size_t f = (size_t)c * NTOK + n;
      ob[f] = g * (o[t][rr] * linv[rr]) + xb[f];
    }
  }
}

// ---------------------------------------------------------------------------
extern "C" void kernel_launch(void* const* d_in, const int* in_sizes, int n_in,
                              void* d_out, int out_size, void* d_ws, size_t ws_size,
                              hipStream_t stream) {
  const float* x     = (const float*)d_in[0];
  const float* Wq    = (const float*)d_in[1];
  const float* bq    = (const float*)d_in[2];
  const float* Wk    = (const float*)d_in[3];
  const float* bk    = (const float*)d_in[4];
  const float* Wv    = (const float*)d_in[5];
  const float* bv    = (const float*)d_in[6];
  const float* gamma = (const float*)d_in[7];
  float* out = (float*)d_out;

  // Workspace carve (bytes):
  char* ws = (char*)d_ws;
  _Float16* QT  = (_Float16*)(ws);                       //  4 MB
  _Float16* KT  = (_Float16*)(ws + 4194304);             //  4 MB
  _Float16* V2  = (_Float16*)(ws + 8388608);             //  8 MB
  _Float16* Xh  = (_Float16*)(ws + 16777216);            //  8 MB
  _Float16* WTq = (_Float16*)(ws + 25165824);            // 64 KB
  _Float16* WTk = (_Float16*)(ws + 25231360);            // 64 KB
  _Float16* WTv = (_Float16*)(ws + 25296896);            // 128 KB

  // Prep: convert x to f16; transpose+convert weights.
  cvt_kernel<<<(NB * NTOK * CIN / 8) / 256, 256, 0, stream>>>(x, Xh, NB * NTOK * CIN / 8);
  wtr_kernel<<<(CH  * CIN) / 256, 256, 0, stream>>>(Wq, WTq, CH);
  wtr_kernel<<<(CH  * CIN) / 256, 256, 0, stream>>>(Wk, WTk, CH);
  wtr_kernel<<<(CIN * CIN) / 256, 256, 0, stream>>>(Wv, WTv, CIN);

  // Projections (16x16 tile per wave, 8 waves/block).
  const int wavesQK = NB * (NTOK / 16) * (CH  / 16);   // 8192
  const int wavesV  = NB * (NTOK / 16) * (CIN / 16);   // 16384
  proj_kernel<<<wavesQK / 8, 256, 0, stream>>>(Xh, WTq, bq, QT, CH, 1);
  proj_kernel<<<wavesQK / 8, 256, 0, stream>>>(Xh, WTk, bk, KT, CH, 1);
  proj_kernel<<<wavesV  / 8, 256, 0, stream>>>(Xh, WTv, bv, V2, CIN, 0);

  // Flash attention: 1024 waves, 128 blocks, 73 KB dynamic LDS each.
  const int wavesA = NB * (NTOK / 16);
  attn_kernel<<<wavesA / WPB, 256, SMEM_BYTES, stream>>>(QT, KT, V2, x, gamma, out);
}
